// GraphAttentionLayer_1640677507273
// MI455X (gfx1250) — matrix operand order repaired and checked by reading, not compiled, not run on hardware
//
#include <hip/hip_runtime.h>
#include <hip/hip_bf16.h>

// GAT forward for MI455X (gfx1250, wave32).
// Kernel 1: h_prime = h x W (bf16 WMMA, fp32 acc) + ei/ej rank-1 reductions.
//           h_prime stored transposed bf16 [b*H+h][f][n] for K-contiguous B-frag loads.
// Kernel 2: flash-style fused mask+leakyrelu+bias+softmax+aggregation, one 16-row
//           i-tile per wave, attention matrix never materialized.

typedef __attribute__((ext_vector_type(16))) __bf16          v16bf;
typedef __attribute__((ext_vector_type(16))) unsigned short  v16u16;
typedef __attribute__((ext_vector_type(8)))  float           v8f;

#define BATCH   8
#define N_NODES 1024
#define FIN     128
#define FOUT    64
#define H_HEADS 4

// ISA 7.12.2 16-bit fragment layouts (wave32):
//  A 16x32 : lane owns row (lane&15); its 16 K-values are the two contiguous
//            8-element runs  kb..kb+7  and  kb+16..kb+23, kb = 8*(lane>=16).
//            Vector element i maps to k = 16*(i>=8) + kb + (i&7).
//  B 32x16 : lane owns col (lane&15); this code loads A and B alike as the contiguous run
//            klo..klo+15, klo = 16*(lane>=16). Vector element i maps to klo+i.
//  C/D f32 : element v maps to (m = v + 8*(lane>=16), n = lane&15).

static __device__ __forceinline__ v16bf cvt16(float4 r0, float4 r1,
                                              float4 r2, float4 r3) {
  v16bf o;
  o[0]  = (__bf16)r0.x; o[1]  = (__bf16)r0.y; o[2]  = (__bf16)r0.z; o[3]  = (__bf16)r0.w;
  o[4]  = (__bf16)r1.x; o[5]  = (__bf16)r1.y; o[6]  = (__bf16)r1.z; o[7]  = (__bf16)r1.w;
  o[8]  = (__bf16)r2.x; o[9]  = (__bf16)r2.y; o[10] = (__bf16)r2.z; o[11] = (__bf16)r2.w;
  o[12] = (__bf16)r3.x; o[13] = (__bf16)r3.y; o[14] = (__bf16)r3.z; o[15] = (__bf16)r3.w;
  return o;
}

// ---------------------------------------------------------------------------
// Kernel 1: projection + attention logit vectors
// grid = 256 blocks x 256 threads (8 waves). 8 blocks per (b,head); each wave
// computes one 16x64 h_prime tile, K = 128 in four bf16-WMMA K-steps.
// ---------------------------------------------------------------------------
__global__ __launch_bounds__(256) void gat_proj(
    const float* __restrict__ h, const float* __restrict__ W,
    const float* __restrict__ a, unsigned short* __restrict__ hp_t,
    float* __restrict__ ei, float* __restrict__ ej)
{
  __shared__ unsigned short w_t[FIN * FOUT];   // W^T bf16: [f][k], 16 KB
  __shared__ float a1s[FOUT], a2s[FOUT];

  const int tid = threadIdx.x;
  const int bh  = blockIdx.x >> 3;             // 8 blocks per (b,head)
  const int b   = bh >> 2;
  const int hh  = bh & 3;

  const float* Wg = W + (size_t)hh * FIN * FOUT;
  for (int idx = tid; idx < FIN * FOUT; idx += 256) {
    int k = idx >> 6, f = idx & 63;            // W[hh][k][f]
    w_t[f * FIN + k] = __builtin_bit_cast(unsigned short, (__bf16)Wg[idx]);
  }
  if (tid < 2 * FOUT) {
    float av = a[hh * 2 * FOUT + tid];
    if (tid < FOUT) a1s[tid] = av; else a2s[tid - FOUT] = av;
  }
  __syncthreads();

  const int lane = tid & 31, wave = tid >> 5;
  const int rt   = (blockIdx.x & 7) * 8 + wave;   // row-tile 0..63
  const int row0 = rt * 16;
  const int m_lo = lane & 15;
  const int hi8  = (lane >> 4) << 3;
  const int koff = (lane >> 4) << 3;              // unused; A-frag loads use klo
  const int klo  = (lane >> 4) << 4;              // B-frag run base

  v8f acc[4] = {};
  const float* hrow = h + ((size_t)b * N_NODES + row0 + m_lo) * FIN;

  for (int ks = 0; ks < FIN; ks += 32) {
    const float* pA = hrow + ks + klo;
    v16bf af = cvt16(*(const float4*)(pA),      *(const float4*)(pA + 4),
                     *(const float4*)(pA + 8),  *(const float4*)(pA + 12));
#pragma unroll
    for (int c = 0; c < 4; ++c) {
      v16u16 bu = *(const v16u16*)(&w_t[(c * 16 + m_lo) * FIN + ks + klo]);
      acc[c] = __builtin_amdgcn_wmma_f32_16x16x32_bf16(
          false, af, false, __builtin_bit_cast(v16bf, bu),
          (short)0, acc[c], false, false);
    }
  }

  // Store h_prime transposed bf16 and fold ei/ej from live accumulators.
  float eiacc[8] = {}, ejacc[8] = {};
#pragma unroll
  for (int c = 0; c < 4; ++c)
#pragma unroll
    for (int v = 0; v < 8; ++v) {
      float val = acc[c][v];
      int f = c * 16 + m_lo;
      int m = v + hi8;
      hp_t[((size_t)bh * FOUT + f) * N_NODES + row0 + m] =
          __builtin_bit_cast(unsigned short, (__bf16)val);
      eiacc[v] += val * a1s[f];
      ejacc[v] += val * a2s[f];
    }
  // Reduce over f (lane index n) within each 16-lane half.
#pragma unroll
  for (int v = 0; v < 8; ++v) {
#pragma unroll
    for (int mask = 1; mask <= 8; mask <<= 1) {
      eiacc[v] += __shfl_xor(eiacc[v], mask, 32);
      ejacc[v] += __shfl_xor(ejacc[v], mask, 32);
    }
  }
  if (lane == 0 || lane == 16) {
#pragma unroll
    for (int v = 0; v < 8; ++v) {
      int r = row0 + v + hi8;
      ei[bh * N_NODES + r] = eiacc[v];
      ej[bh * N_NODES + r] = ejacc[v];
    }
  }
}

// ---------------------------------------------------------------------------
// Kernel 2: fused attention + aggregation. One wave = one (b,head,16-row tile),
// streaming j in K=32 chunks; P built directly in WMMA A-layout; 4 WMMA per
// chunk accumulate the 16x64 output in fp32.
// ---------------------------------------------------------------------------
__global__ __launch_bounds__(256) void gat_attn(
    const int* __restrict__ adj, const float* __restrict__ bias,
    const unsigned short* __restrict__ hp_t,
    const float* __restrict__ ei, const float* __restrict__ ej,
    float* __restrict__ out)
{
  const int tid  = threadIdx.x, lane = tid & 31, wave = tid >> 5;
  const int tile = blockIdx.x * 8 + wave;       // 0..2047
  const int bh   = tile >> 6, rt = tile & 63;
  const int b    = bh >> 2,  hh = bh & 3;
  const int row0 = rt * 16;
  const int m_lo = lane & 15;
  const int hi8  = (lane >> 4) << 3;
  const int koff = (lane >> 4) << 3;            // unused; A-frag loads use klo
  const int klo  = (lane >> 4) << 4;            // B-frag run base
  const int i_ln = row0 + m_lo;                 // P-row owned by this lane

  const float  ei_m   = ei[bh * N_NODES + i_ln];
  const float* ejp    = ej + bh * N_NODES;
  const float* biasRw = bias + (size_t)i_ln * N_NODES;
  const int*   adjRw  = adj  + (size_t)i_ln * N_NODES;
  const unsigned short* hpb = hp_t + (size_t)bh * FOUT * N_NODES;

  v8f   acc[4] = {};
  float sumloc = 0.f;

  for (int j0 = 0; j0 < N_NODES; j0 += 32) {
    if (j0 + 32 < N_NODES) {
      __builtin_prefetch(biasRw + j0 + 32, 0, 1);
      __builtin_prefetch(adjRw  + j0 + 32, 0, 1);
    }
    // One contiguous 16-element run per lane: [j0+klo, +16)
    const int jb = j0 + klo;
    float4 ej0 = *(const float4*)(ejp + jb);
    float4 ej1 = *(const float4*)(ejp + jb + 4);
    float4 ej2 = *(const float4*)(ejp + jb + 8);
    float4 ej3 = *(const float4*)(ejp + jb + 12);
    float4 bv0 = *(const float4*)(biasRw + jb);
    float4 bv1 = *(const float4*)(biasRw + jb + 4);
    float4 bv2 = *(const float4*)(biasRw + jb + 8);
    float4 bv3 = *(const float4*)(biasRw + jb + 12);
    int4   av0 = *(const int4*)(adjRw + jb);
    int4   av1 = *(const int4*)(adjRw + jb + 4);
    int4   av2 = *(const int4*)(adjRw + jb + 8);
    int4   av3 = *(const int4*)(adjRw + jb + 12);

    float p[16];
#pragma unroll
    for (int q = 0; q < 16; ++q) {
      float ejv = ((const float*)&ej0)[q < 4 ? q : 100];  // placeholder, replaced below
      (void)ejv;
    }
    // compute 16 probabilities (runs flattened in vector-element order)
    {
      const float* ejr = (const float*)&ej0;   // ej0..ej3 are contiguous on stack
      const float* bvr = (const float*)&bv0;
      const int*   avr = (const int*)&av0;
#pragma unroll
      for (int q = 0; q < 16; ++q) {
        float e = ei_m + ejr[q];
        e = (e > 0.f) ? e : 0.2f * e;          // leaky_relu(0.2)
        e += bvr[q];
        p[q] = avr[q] ? __expf(e) : 0.f;       // masked -> exact 0
      }
    }
    float s = 0.f;
#pragma unroll
    for (int q = 0; q < 16; ++q) s += p[q];
    sumloc += s;

    v16bf pf = cvt16(make_float4(p[0],  p[1],  p[2],  p[3]),
                     make_float4(p[4],  p[5],  p[6],  p[7]),
                     make_float4(p[8],  p[9],  p[10], p[11]),
                     make_float4(p[12], p[13], p[14], p[15]));
#pragma unroll
    for (int c = 0; c < 4; ++c) {
      v16u16 bu = *(const v16u16*)(hpb + (size_t)(c * 16 + m_lo) * N_NODES + j0 + klo);
      acc[c] = __builtin_amdgcn_wmma_f32_16x16x32_bf16(
          false, pf, false, __builtin_bit_cast(v16bf, bu),
          (short)0, acc[c], false, false);
    }
  }

  // Row m lives in lanes m and m^16: combine, then broadcast per output row.
  float rowsum = sumloc + __shfl_xor(sumloc, 16, 32);
#pragma unroll
  for (int v = 0; v < 8; ++v) {
    int   m   = v + hi8;
    float inv = 1.0f / __shfl(rowsum, m, 32);
    float* orow = out + ((size_t)b * N_NODES + row0 + m) * (H_HEADS * FOUT)
                      + hh * FOUT + m_lo;
#pragma unroll
    for (int c = 0; c < 4; ++c)
      orow[c * 16] = acc[c][v] * inv;
  }
}

// ---------------------------------------------------------------------------
extern "C" void kernel_launch(void* const* d_in, const int* in_sizes, int n_in,
                              void* d_out, int out_size, void* d_ws, size_t ws_size,
                              hipStream_t stream) {
  const float* h    = (const float*)d_in[0];
  const int*   adj  = (const int*)  d_in[1];
  const float* bias = (const float*)d_in[2];
  const float* W    = (const float*)d_in[3];
  const float* a    = (const float*)d_in[4];
  float*       out  = (float*)d_out;

  // Workspace: hp_t bf16 [B*H][FOUT][N] (4 MB), then ei/ej fp32 [B*H][N].
  unsigned short* hp_t = (unsigned short*)d_ws;
  size_t hp_bytes = (size_t)BATCH * H_HEADS * FOUT * N_NODES * sizeof(unsigned short);
  float* ei = (float*)((char*)d_ws + hp_bytes);
  float* ej = ei + (size_t)BATCH * H_HEADS * N_NODES;

  gat_proj<<<256, 256, 0, stream>>>(h, W, a, hp_t, ei, ej);
  gat_attn<<<256, 256, 0, stream>>>(adj, bias, hp_t, ei, ej, out);
}
